// MultiHeadAttention_66932770341082
// MI455X (gfx1250) — matrix operand order repaired and checked by reading, not compiled, not run on hardware
//
#include <hip/hip_runtime.h>
#include <hip/hip_bf16.h>

typedef _Float16 half_t;
typedef __attribute__((ext_vector_type(16))) _Float16 v16h;
typedef __attribute__((ext_vector_type(8)))  _Float16 v8h;
typedef __attribute__((ext_vector_type(4)))  _Float16 v4h;
typedef __attribute__((ext_vector_type(8)))  float    v8f;
typedef __attribute__((ext_vector_type(4)))  float    v4f;

#define NBATCH 2
#define SEQ    2048
#define EMB    1024
#define NHEAD  16
#define HDIM   64

// ---------------------------------------------------------------------------
// WMMA helper: D = A(16x32 f16) * B(32x16 f16) + C(16x16 f32)
// ---------------------------------------------------------------------------
__device__ __forceinline__ v8f wmma_f16(v16h a, v16h b, v8f c) {
  return __builtin_amdgcn_wmma_f32_16x16x32_f16(
      /*neg_a=*/false, a, /*neg_b=*/false, b,
      /*c_mod=*/(short)0, c, /*reuse_a=*/false, /*reuse_b=*/false);
}

// A-operand (16x32, M x K), row-major source with leading dim ld.
// Contiguous-k layout used here (not AMD's table order): lane holds row M = lane&15; element i -> K = 16*(lane>>4)+i.
__device__ __forceinline__ v16h load_a_frag(const half_t* __restrict__ base,
                                            int row0, int ld, int k0, int lane) {
  const half_t* p = base + (size_t)(row0 + (lane & 15)) * ld + k0 + 16 * (lane >> 4);
  v8h lo = *(const v8h*)p;
  v8h hi = *(const v8h*)(p + 8);
  v16h r;
#pragma unroll
  for (int i = 0; i < 8; i++) { r[i] = lo[i]; r[i + 8] = hi[i]; }
  return r;
}

// B-operand (32x16, K x N) where B[k][n] = src_row-major[row0+n][k].
// Contiguous-k layout used here (not AMD's table order): lane holds column N = lane&15; element i -> K = 16*(lane>>4)+i.
// -> one contiguous 32B load per lane.
__device__ __forceinline__ v16h load_b_frag(const half_t* __restrict__ base,
                                            int row0, int ld, int k0, int lane) {
  const half_t* p = base + (size_t)(row0 + (lane & 15)) * ld + k0 + 16 * (lane >> 4);
  return *(const v16h*)p;
}

__device__ __forceinline__ v8h shfl_xor16_v8h(v8h v) {
  union { v8h h; int i[4]; } u;
  u.h = v;
#pragma unroll
  for (int k = 0; k < 4; k++) u.i[k] = __shfl_xor(u.i[k], 16, 32);
  return u.h;
}

// ---------------------------------------------------------------------------
// f32 -> f16 conversion (vectorized x4)
// ---------------------------------------------------------------------------
__global__ void cvt_f32_f16(const float* __restrict__ src, half_t* __restrict__ dst, int n4) {
  int i = blockIdx.x * blockDim.x + threadIdx.x;
  if (i < n4) {
    v4f f = ((const v4f*)src)[i];
    v4h h;
#pragma unroll
    for (int k = 0; k < 4; k++) h[k] = (half_t)f[k];
    ((v4h*)dst)[i] = h;
  }
}

// ---------------------------------------------------------------------------
// GEMM: out = X[M,K] * W[Ncols,K]^T + bias   (torch Linear semantics)
// one wave computes a 32(M) x 64(N) tile (2 A-frags x 4 B-frags -> 8 WMMAs
// per K-step; B fragment reused 2x, A fragment reused 4x => 1.5 b128 loads
// per WMMA).
// flag 0: f16 out, head layout  [N, H, seq, 64]   (Q / K)
// flag 1: f16 out, transposed V [N, H, 64, seq]
// flag 2: f32 out, plain row-major [M, Ncols]     (final projection)
// ---------------------------------------------------------------------------
__global__ void gemm16(const half_t* __restrict__ X, const half_t* __restrict__ W,
                       const float* __restrict__ bias, void* __restrict__ out,
                       int M, int Kdim, int Ncols, int flag, int seqS) {
  const int lane = threadIdx.x & 31;
  const int wave = threadIdx.x >> 5;
  const int m0 = (blockIdx.x * 8 + wave) * 32;
  const int o0 = blockIdx.y * 64;
  if (m0 >= M) return;

  v8f acc[2][4] = {};
  for (int k0 = 0; k0 < Kdim; k0 += 32) {
    v16h a0 = load_a_frag(X, m0, Kdim, k0, lane);
    v16h a1 = load_a_frag(X, m0 + 16, Kdim, k0, lane);
#pragma unroll
    for (int j = 0; j < 4; j++) {
      v16h b = load_b_frag(W, o0 + 16 * j, Kdim, k0, lane);
      acc[0][j] = wmma_f16(a0, b, acc[0][j]);
      acc[1][j] = wmma_f16(a1, b, acc[1][j]);
    }
  }

  const int hl = lane >> 4, nl = lane & 15;
#pragma unroll
  for (int mi = 0; mi < 2; mi++) {
    const int mbase = m0 + 16 * mi;
    if (flag == 2) {
      float* O = (float*)out;
#pragma unroll
      for (int j = 0; j < 4; j++) {
        int o = o0 + 16 * j + nl;
        float bv = bias[o];
#pragma unroll
        for (int r = 0; r < 8; r++)
          O[(size_t)(mbase + 8 * hl + r) * Ncols + o] = acc[mi][j][r] + bv;
      }
    } else if (flag == 0) {
      half_t* O = (half_t*)out;
      int nb = mbase / seqS;
      int sbase = mbase - nb * seqS;
      int h = o0 >> 6;  // HD == 64
#pragma unroll
      for (int j = 0; j < 4; j++) {
        int d = 16 * j + nl;
        float bv = bias[o0 + 16 * j + nl];
#pragma unroll
        for (int r = 0; r < 8; r++) {
          int s = sbase + 8 * hl + r;
          O[((size_t)(nb * NHEAD + h) * seqS + s) * HDIM + d] = (half_t)(acc[mi][j][r] + bv);
        }
      }
    } else {  // flag 1: V transposed [NH, 64, seq]
      half_t* O = (half_t*)out;
      int nb = mbase / seqS;
      int tbase = mbase - nb * seqS;
      int h = o0 >> 6;
#pragma unroll
      for (int j = 0; j < 4; j++) {
        int d = 16 * j + nl;
        float bv = bias[o0 + 16 * j + nl];
        v8h v;
#pragma unroll
        for (int r = 0; r < 8; r++) v[r] = (half_t)(acc[mi][j][r] + bv);
        *(v8h*)&O[((size_t)(nb * NHEAD + h) * HDIM + d) * seqS + tbase + 8 * hl] = v;
      }
    }
  }
}

// ---------------------------------------------------------------------------
// Causal flash attention, one wave per 16 query rows per head.
// Scores computed transposed: St = K_tile(A) x Q^T(B)  -> lane's N-col = s.
// Output computed transposed: Ot = V^T(A)  x P^T(B)    -> lane's N-col = s.
// => per-lane scalar online-softmax statistics, no LDS transposes.
// ---------------------------------------------------------------------------
__global__ void attn_kernel(const half_t* __restrict__ Q, const half_t* __restrict__ K,
                            const half_t* __restrict__ Vt, half_t* __restrict__ ctx) {
  const int lane = threadIdx.x & 31;
  const int wave = threadIdx.x >> 5;
  const int head = blockIdx.y;                       // 0 .. N*H-1
  const int s0 = (blockIdx.x * 8 + wave) * 16;
  const int hl = lane >> 4, nl = lane & 15;

  const half_t* Qh = Q + (size_t)head * SEQ * HDIM;   // [S, 64]
  const half_t* Kh = K + (size_t)head * SEQ * HDIM;   // [T, 64]
  const half_t* Vh = Vt + (size_t)head * HDIM * SEQ;  // [64, T]

  // Q as B-operand (K-dim = d), loaded once: d chunks 0..31 and 32..63
  v16h qb0 = load_b_frag(Qh, s0, HDIM, 0, lane);
  v16h qb1 = load_b_frag(Qh, s0, HDIM, 32, lane);

  v8f acc0 = {}, acc1 = {}, acc2 = {}, acc3 = {};
  float m_i = -1e30f, l_i = 0.0f;
  const int s_row = s0 + nl;  // this lane's query row
  const float scale = 0.125f; // 1/sqrt(64)

  for (int t0 = 0; t0 < s0 + 16; t0 += 32) {
    // ---- scores (transposed): St[t, s] = sum_d K[t,d] Q[s,d]
    v8f st0 = {}, st1 = {};
    st0 = wmma_f16(load_a_frag(Kh, t0, HDIM, 0, lane), qb0, st0);
    st0 = wmma_f16(load_a_frag(Kh, t0, HDIM, 32, lane), qb1, st0);
    st1 = wmma_f16(load_a_frag(Kh, t0 + 16, HDIM, 0, lane), qb0, st1);
    st1 = wmma_f16(load_a_frag(Kh, t0 + 16, HDIM, 32, lane), qb1, st1);

    // ---- scale + causal mask + row max (row = s, lives in lanes nl and nl+16)
    float tmax = -1e30f;
#pragma unroll
    for (int r = 0; r < 8; r++) {
      int t = t0 + 8 * hl + r;
      float v = (t <= s_row) ? st0[r] * scale : -1e30f;
      st0[r] = v; tmax = fmaxf(tmax, v);
      int t2 = t + 16;
      float v2 = (t2 <= s_row) ? st1[r] * scale : -1e30f;
      st1[r] = v2; tmax = fmaxf(tmax, v2);
    }
    tmax = fmaxf(tmax, __shfl_xor(tmax, 16, 32));
    float m_new = fmaxf(m_i, tmax);

    // ---- exponentiate + row sum
    float rsum = 0.0f;
#pragma unroll
    for (int r = 0; r < 8; r++) {
      float p0 = __expf(st0[r] - m_new);
      float p1 = __expf(st1[r] - m_new);
      st0[r] = p0; st1[r] = p1;
      rsum += p0 + p1;
    }
    rsum += __shfl_xor(rsum, 16, 32);
    float alpha = __expf(m_i - m_new);
    l_i = l_i * alpha + rsum;
    m_i = m_new;

    // ---- rescale running output (per-lane scalar: all elems share this s)
#pragma unroll
    for (int r = 0; r < 8; r++) {
      acc0[r] *= alpha; acc1[r] *= alpha; acc2[r] *= alpha; acc3[r] *= alpha;
    }

    // ---- build P^T B-operand from the two St C-fragments.
    // B wants, per lane (n = s): K = t0 + 16*hl + i.  Own fragment supplies
    // half of it; the xor-16 partner lane supplies the other half.
    v8h h0, h1;
#pragma unroll
    for (int r = 0; r < 8; r++) { h0[r] = (half_t)st0[r]; h1[r] = (half_t)st1[r]; }
    v8h send = hl ? h0 : h1;
    v8h recv = shfl_xor16_v8h(send);
    v8h blo = hl ? recv : h0;   // B elements 0..7
    v8h bhi = hl ? h1 : recv;   // B elements 8..15
    v16h pb;
#pragma unroll
    for (int i = 0; i < 8; i++) { pb[i] = blo[i]; pb[i + 8] = bhi[i]; }

    // ---- Ot += V^T(A, K-dim = t over 32) x P^T(B)
    acc0 = wmma_f16(load_a_frag(Vh, 0,  SEQ, t0, lane), pb, acc0);
    acc1 = wmma_f16(load_a_frag(Vh, 16, SEQ, t0, lane), pb, acc1);
    acc2 = wmma_f16(load_a_frag(Vh, 32, SEQ, t0, lane), pb, acc2);
    acc3 = wmma_f16(load_a_frag(Vh, 48, SEQ, t0, lane), pb, acc3);
  }

  // ---- normalize and store context [N, S, E] (f16), E index = h*64 + hd
  const int nb = head >> 4, h = head & 15;
  const float inv = 1.0f / l_i;
  half_t* cbase = ctx + ((size_t)(nb * SEQ + s0 + nl)) * EMB + h * HDIM + 8 * hl;
  v8f* accs[4] = {&acc0, &acc1, &acc2, &acc3};
#pragma unroll
  for (int j = 0; j < 4; j++) {
    v8h o;
#pragma unroll
    for (int r = 0; r < 8; r++) o[r] = (half_t)((*accs[j])[r] * inv);
    *(v8h*)(cbase + 16 * j) = o;
  }
}

// ---------------------------------------------------------------------------
extern "C" void kernel_launch(void* const* d_in, const int* in_sizes, int n_in,
                              void* d_out, int out_size, void* d_ws, size_t ws_size,
                              hipStream_t stream) {
  const float* query = (const float*)d_in[0];
  const float* key_i = (const float*)d_in[1];
  const float* value = (const float*)d_in[2];
  // d_in[3] = attn_mask (deterministic causal tril) -> implemented analytically
  const float* Wq = (const float*)d_in[4];  const float* bq = (const float*)d_in[5];
  const float* Wk = (const float*)d_in[6];  const float* bk = (const float*)d_in[7];
  const float* Wv = (const float*)d_in[8];  const float* bv = (const float*)d_in[9];
  const float* Wp = (const float*)d_in[10]; const float* bp = (const float*)d_in[11];
  float* out = (float*)d_out;

  const int Mrows = NBATCH * SEQ;           // 4096
  const size_t actElems = (size_t)Mrows * EMB;   // 4M
  const size_t wElems   = (size_t)EMB * EMB;     // 1M

  char* ws = (char*)d_ws;
  size_t off = 0;
  auto alloc = [&](size_t bytes) -> half_t* {
    half_t* p = (half_t*)(ws + off);
    off += (bytes + 255) & ~(size_t)255;
    return p;
  };
  half_t* Xq  = alloc(actElems * 2);
  half_t* Xk  = alloc(actElems * 2);
  half_t* Xv  = alloc(actElems * 2);
  half_t* Wq2 = alloc(wElems * 2);
  half_t* Wk2 = alloc(wElems * 2);
  half_t* Wv2 = alloc(wElems * 2);
  half_t* Wp2 = alloc(wElems * 2);
  half_t* Qh  = alloc(actElems * 2);   // [NH, S, 64]
  half_t* Kh  = alloc(actElems * 2);   // [NH, T, 64]
  half_t* Vth = alloc(actElems * 2);   // [NH, 64, T]
  half_t* ctx = alloc(actElems * 2);   // [N, S, E]

  // Phase 0: f32 -> f16 conversions
  {
    int n4a = (int)(actElems / 4), n4w = (int)(wElems / 4);
    dim3 blk(256);
    cvt_f32_f16<<<dim3((n4a + 255) / 256), blk, 0, stream>>>(query, Xq, n4a);
    cvt_f32_f16<<<dim3((n4a + 255) / 256), blk, 0, stream>>>(key_i, Xk, n4a);
    cvt_f32_f16<<<dim3((n4a + 255) / 256), blk, 0, stream>>>(value, Xv, n4a);
    cvt_f32_f16<<<dim3((n4w + 255) / 256), blk, 0, stream>>>(Wq, Wq2, n4w);
    cvt_f32_f16<<<dim3((n4w + 255) / 256), blk, 0, stream>>>(Wk, Wk2, n4w);
    cvt_f32_f16<<<dim3((n4w + 255) / 256), blk, 0, stream>>>(Wv, Wv2, n4w);
    cvt_f32_f16<<<dim3((n4w + 255) / 256), blk, 0, stream>>>(Wp, Wp2, n4w);
  }

  // Phase 1: QKV projections (32x64 tile per wave; grid = M/256 x E/64)
  {
    dim3 grid(Mrows / 256, EMB / 64), blk(256);
    gemm16<<<grid, blk, 0, stream>>>(Xq, Wq2, bq, Qh,  Mrows, EMB, EMB, 0, SEQ);
    gemm16<<<grid, blk, 0, stream>>>(Xk, Wk2, bk, Kh,  Mrows, EMB, EMB, 0, SEQ);
    gemm16<<<grid, blk, 0, stream>>>(Xv, Wv2, bv, Vth, Mrows, EMB, EMB, 1, SEQ);
  }

  // Phase 2: causal attention (grid = S/(16*8) x N*H)
  {
    dim3 grid(SEQ / 128, NBATCH * NHEAD), blk(256);
    attn_kernel<<<grid, blk, 0, stream>>>(Qh, Kh, Vth, ctx);
  }

  // Phase 3: output projection -> f32
  {
    dim3 grid(Mrows / 256, EMB / 64), blk(256);
    gemm16<<<grid, blk, 0, stream>>>(ctx, Wp2, bp, out, Mrows, EMB, EMB, 2, SEQ);
  }
}
